// GPT2SoftmaxForNer_LE_12206297055419
// MI455X (gfx1250) — compile-verified, matrix operand order checked by reading, not run on hardware
//
#include <hip/hip_runtime.h>
#include <hip/hip_bf16.h>

// Shapes (fixed by the reference)
#define BB 16
#define SS 512
#define HH 768
#define EE 19
#define T0 3
#define MTOK (BB * SS)      // 8192 tokens
#define MTILE 16            // tokens per workgroup
#define NWAVE 8             // waves per workgroup (wave32)
#define HPL (HH / 32)       // 24 h-elements per lane in phase B

typedef __attribute__((ext_vector_type(2)))  float  v2f;
typedef __attribute__((ext_vector_type(8)))  float  v8f;
typedef __attribute__((ext_vector_type(16))) __bf16 v16bf;

__global__ __launch_bounds__(256) void ner_le_fused_kernel(
    const float* __restrict__ seq,      // [B,S,H]
    const float* __restrict__ lab,      // [E,H]
    const float* __restrict__ attn_w,   // [H,H] row-major; x_attn = seq @ W^T
    const float* __restrict__ attn_b,   // [H]
    const float* __restrict__ fc_w,     // [H]
    const int*   __restrict__ counts,   // [B]
    float* __restrict__ out)            // [B,S,H]
{
    __shared__ float lds_A[MTILE * HH]; // seq tile, fp32 (48 KB)
    __shared__ float lds_X[MTILE * HH]; // x_attn tile, fp32 (48 KB)

    const int wg   = blockIdx.x;        // 0..511
    const int m0   = wg * MTILE;        // first token of tile
    const int tid  = threadIdx.x;
    const int lane = tid & 31;
    const int wave = tid >> 5;

    // ---- load 16x768 seq tile into LDS (coalesced float4) ----
    {
        const float4* g = (const float4*)(seq + (size_t)m0 * HH);
        float4* s = (float4*)lds_A;
        for (int i = tid; i < MTILE * HH / 4; i += 256) s[i] = g[i];
    }
    __syncthreads();

    // ---- Phase A: x_attn tile = A(16x768) * W^T, via fp32 WMMA ----
    // Each wave owns 96 columns = 6 N-tiles, processed as 3 pairs so two
    // independent accumulator chains interleave on the matrix pipe and one
    // A-fragment LDS read feeds two WMMAs.
    const int row = lane & 15;          // A row / B column / C column within 16-tile

    for (int np = 0; np < 3; ++np) {
        const int n0 = wave * 96 + np * 32;                 // first column of the pair
        const float* __restrict__ bcol0 = attn_w + (size_t)(n0 + row) * HH;
        const float* __restrict__ bcol1 = bcol0 + 16 * HH;  // second tile's column
        const float* __restrict__ arow  = lds_A + row * HH;
        __builtin_prefetch(bcol0, 0, 0);                    // global_prefetch_b8
        __builtin_prefetch(bcol1, 0, 0);

        v8f acc0 = {};
        v8f acc1 = {};
#if __has_builtin(__builtin_amdgcn_wmma_f32_16x16x4_f32)
        // Full-precision fp32 matrix pipe: 16x16x4, K swept in steps of 4.
        const int kq2 = (lane >> 4) * 2;                    // lanes 0-15: K{0,1}; 16-31: K{2,3}
        for (int k0 = 0; k0 < HH; k0 += 4) {
            v2f a  = *(const v2f*)(arow  + k0 + kq2);
            v2f b0 = *(const v2f*)(bcol0 + k0 + kq2);
            v2f b1 = *(const v2f*)(bcol1 + k0 + kq2);
            acc0 = __builtin_amdgcn_wmma_f32_16x16x4_f32(
                false, a, false, b0, (short)0, acc0, false, false);
            acc1 = __builtin_amdgcn_wmma_f32_16x16x4_f32(
                false, a, false, b1, (short)0, acc1, false, false);
        }
#else
        // Fallback: bf16 WMMA (convert fp32 -> bf16 on the fly)
        const int kb1 = (lane < 16) ? 0 : 8;
        const int kb2 = kb1 + 16;
        for (int k0 = 0; k0 < HH; k0 += 32) {
            v16bf a, b0, b1;
#pragma unroll
            for (int j = 0; j < 8; ++j) {
                a[j]      = (__bf16)arow[k0 + kb1 + j];
                a[j + 8]  = (__bf16)arow[k0 + kb2 + j];
                b0[j]     = (__bf16)bcol0[k0 + kb1 + j];
                b0[j + 8] = (__bf16)bcol0[k0 + kb2 + j];
                b1[j]     = (__bf16)bcol1[k0 + kb1 + j];
                b1[j + 8] = (__bf16)bcol1[k0 + kb2 + j];
            }
            acc0 = __builtin_amdgcn_wmma_f32_16x16x32_bf16(
                false, a, false, b0, (short)0, acc0, false, false);
            acc1 = __builtin_amdgcn_wmma_f32_16x16x32_bf16(
                false, a, false, b1, (short)0, acc1, false, false);
        }
#endif
        // C/D layout: VGPR r -> (M = r + 8*(lane>=16), N = lane&15)
        const int mbase = (lane >> 4) * 8;
        const int ncol0 = n0 + row;
        const int ncol1 = ncol0 + 16;
        const float bias0 = attn_b[ncol0];
        const float bias1 = attn_b[ncol1];
#pragma unroll
        for (int r = 0; r < 8; ++r) {
            lds_X[(mbase + r) * HH + ncol0] = acc0[r] + bias0;
            lds_X[(mbase + r) * HH + ncol1] = acc1[r] + bias1;
        }
    }
    __syncthreads();

    // ---- Phase B: fused tanh-dot / softmax / context / residual ----
    float fw[HPL];
#pragma unroll
    for (int j = 0; j < HPL; ++j) fw[j] = fc_w[lane + 32 * j];

    for (int t = wave; t < MTILE; t += NWAVE) {             // 2 tokens per wave
        const int gtok = m0 + t;
        const int bidx = gtok >> 9;                          // / SS
        const int sidx = gtok & (SS - 1);
        const bool keep = (sidx >= T0) && (sidx < counts[bidx] + T0);

        float xv[HPL], sv[HPL];
#pragma unroll
        for (int j = 0; j < HPL; ++j) {
            const int h = lane + 32 * j;
            xv[j] = lds_X[t * HH + h];
            sv[j] = lds_A[t * HH + h];
        }

        // scores[e] = sum_h tanh(x_attn + lab[e,h]) * fc_w[h]
        float p[EE];
        float mx = -3.4e38f;
#pragma unroll 1
        for (int e = 0; e < EE; ++e) {
            const float* __restrict__ le = lab + e * HH;
            float part = 0.f;
#pragma unroll
            for (int j = 0; j < HPL; ++j)
                part += tanhf(xv[j] + le[lane + 32 * j]) * fw[j];
#pragma unroll
            for (int off = 16; off; off >>= 1)
                part += __shfl_xor(part, off, 32);
            p[e] = part;
            mx = fmaxf(mx, part);
        }

        // softmax over E (redundant per lane, all lanes hold full vector)
        float sum = 0.f;
#pragma unroll
        for (int e = 0; e < EE; ++e) { p[e] = __expf(p[e] - mx); sum += p[e]; }
        const float inv = 1.f / sum;
#pragma unroll
        for (int e = 0; e < EE; ++e) p[e] *= inv;

        // out = keep ? seq : (seq + p @ lab)
        float* __restrict__ op = out + (size_t)gtok * HH;
#pragma unroll 4
        for (int j = 0; j < HPL; ++j) {
            const int h = lane + 32 * j;
            float ctx = 0.f;
#pragma unroll
            for (int e = 0; e < EE; ++e) ctx += p[e] * lab[e * HH + h];
            op[h] = keep ? sv[j] : (sv[j] + ctx);
        }
    }
}

extern "C" void kernel_launch(void* const* d_in, const int* in_sizes, int n_in,
                              void* d_out, int out_size, void* d_ws, size_t ws_size,
                              hipStream_t stream) {
    (void)in_sizes; (void)n_in; (void)out_size; (void)d_ws; (void)ws_size;
    const float* seq    = (const float*)d_in[0];
    const float* lab    = (const float*)d_in[1];
    const float* attn_w = (const float*)d_in[2];
    const float* attn_b = (const float*)d_in[3];
    const float* fc_w   = (const float*)d_in[4];
    const int*   counts = (const int*)d_in[5];
    float* out = (float*)d_out;

    dim3 grid(MTOK / MTILE);   // 512 workgroups
    dim3 block(256);           // 8 wave32's
    hipLaunchKernelGGL(ner_le_fused_kernel, grid, block, 0, stream,
                       seq, lab, attn_w, attn_b, fc_w, counts, out);
}
